// RCT_8675833938322
// MI455X (gfx1250) — matrix operand order repaired and checked by reading, not compiled, not run on hardware
//
#include <hip/hip_runtime.h>

typedef unsigned short u16;
typedef __attribute__((ext_vector_type(16))) __bf16 v16bf;
typedef __attribute__((ext_vector_type(8)))  float  v8f;
typedef __attribute__((ext_vector_type(4)))  unsigned int u32x4;
typedef __attribute__((ext_vector_type(8)))  int i32x8;
typedef __attribute__((ext_vector_type(4)))  int i32x4;
typedef __attribute__((address_space(3))) u16 lds16_t;

#if defined(__HIP_DEVICE_COMPILE__)
#if __has_builtin(__builtin_amdgcn_tensor_load_to_lds) && __has_builtin(__builtin_amdgcn_s_wait_tensorcnt)
#define USE_TDM 1
#else
#define USE_TDM 0
#endif
#else
#define USE_TDM 0
#endif

// ---------------- helpers ----------------

__device__ __forceinline__ u16 f2bf(float x) {
    unsigned u = __float_as_uint(x);
    u += 0x7fffu + ((u >> 16) & 1u);
    return (u16)(u >> 16);
}

// A-fragment (16x32, 16-bit): lane = M row; lanes 0-15 hold K {0..7,16..23},
// lanes 16-31 hold K {8..15,24..31} (two contiguous 16B chunks per lane).
__device__ __forceinline__ v16bf load_a_frag(const u16* p, int row0, int k0, int ld) {
    int lane = threadIdx.x & 31;
    int r  = row0 + (lane & 15);
    int kb = k0 + ((lane >> 4) << 3);
    const u16* base = p + (size_t)r * ld + kb;
    v16bf f;
    ((uint4*)&f)[0] = ((const uint4*)base)[0];
    ((uint4*)&f)[1] = ((const uint4*)(base + 16))[0];
    return f;
}

// B-fragment (32x16, 16-bit): lane = N column; lanes 0-15 hold K {0..7,16..23},
// lanes 16-31 hold K {8..15,24..31} (two contiguous 16B chunks per lane).
// Source matrix stored [N,K] row-major (NT gemm / K^T for attention scores).
__device__ __forceinline__ v16bf load_b_frag(const u16* p, int col0, int k0, int ld) {
    int lane = threadIdx.x & 31;
    int r  = col0 + (lane & 15);
    int kb = k0 + ((lane >> 4) << 3);
    const u16* base = p + (size_t)r * ld + kb;
    v16bf f;
    ((uint4*)&f)[0] = ((const uint4*)base)[0];
    ((uint4*)&f)[1] = ((const uint4*)(base + 16))[0];
    return f;
}

// ---------------- fp32 -> bf16 conversion ----------------

__global__ void cvt_bf16(const float* __restrict__ in, u16* __restrict__ out, int n) {
    int i = blockIdx.x * 256 + threadIdx.x;
    int stride = gridDim.x * 256;
    for (; i < n; i += stride) out[i] = f2bf(in[i]);
}

// ---------------- NT GEMM: C[M,N] = A[M,K] * B[N,K]^T ----------------
// OUTMODE: 0 = f32 out, 1 = bf16 out, 2 = relu + bf16 out
// Register double-buffered: fragments for k+32 load while k's WMMAs run.
template <int OUTMODE>
__global__ __launch_bounds__(256)
void gemm_nt(const u16* __restrict__ A, const u16* __restrict__ B,
             void* __restrict__ C, int M, int N, int K) {
    int wave = threadIdx.x >> 5;
    int lane = threadIdx.x & 31;
    int wm = wave >> 1, wn = wave & 1;
    int mBase = blockIdx.y * 128 + wm * 32;
    int nBase = blockIdx.x * 128 + wn * 64;

    v16bf af[2][2];
    v16bf bfr[2][4];
    auto loadTiles = [&](int buf, int k0) {
        af[buf][0] = load_a_frag(A, mBase,      k0, K);
        af[buf][1] = load_a_frag(A, mBase + 16, k0, K);
#pragma unroll
        for (int j = 0; j < 4; ++j) bfr[buf][j] = load_b_frag(B, nBase + j * 16, k0, K);
    };

    v8f acc[2][4] = {};
    loadTiles(0, 0);
    int cur = 0;
    for (int k0 = 0; k0 < K; k0 += 32) {
        int nk = k0 + 32;
        if (nk < K) loadTiles(cur ^ 1, nk);            // overlap with WMMAs below
        if (nk + 32 < K) {                             // L2 prefetch two steps ahead
            __builtin_prefetch(A + (size_t)(mBase + (lane & 15)) * K + nk + 32, 0, 1);
            __builtin_prefetch(B + (size_t)(nBase + (lane & 15)) * K + nk + 32, 0, 1);
        }
#pragma unroll
        for (int i = 0; i < 2; ++i)
#pragma unroll
            for (int j = 0; j < 4; ++j)
                acc[i][j] = __builtin_amdgcn_wmma_f32_16x16x32_bf16(
                    false, af[cur][i], false, bfr[cur][j], (short)0, acc[i][j], false, false);
        cur ^= 1;
    }

    int hi8 = (lane >> 4) << 3;
    int col = lane & 15;
#pragma unroll
    for (int i = 0; i < 2; ++i)
#pragma unroll
        for (int j = 0; j < 4; ++j)
#pragma unroll
            for (int r = 0; r < 8; ++r) {
                int m = mBase + i * 16 + r + hi8;
                int n = nBase + j * 16 + col;
                float v = acc[i][j][r];
                if (OUTMODE == 0) {
                    ((float*)C)[(size_t)m * N + n] = v;
                } else {
                    if (OUTMODE == 2) v = fmaxf(v, 0.0f);
                    ((u16*)C)[(size_t)m * N + n] = f2bf(v);
                }
            }
}

// ---------------- flash attention (B=4, N=2048, D=512, H=2, gs=256) -----
// Block = 128 threads (4 waves) share one (b,h); waves differ in query rows.
// K tile (32x256 bf16, strided rows) is staged by the Tensor Data Mover
// (tensor_load_to_lds, waited with s_wait_tensorcnt) when available; V tile
// is staged transposed by hand so P@V B-fragments are contiguous ds_load_b128.
__global__ __launch_bounds__(128)
void attn_flash(const u16* __restrict__ Qb, const u16* __restrict__ Kb,
                const u16* __restrict__ Vb, float* __restrict__ Out) {
    constexpr int N = 2048, D = 512, GS = 256;
    __shared__ u16 lds_k[32 * 256];        // [kv][d]   16 KB
    __shared__ u16 lds_v[256 * 32];        // [d][kv]   16 KB (transposed)
    __shared__ u16 pbuf[4][16 * 32];       // per-wave P tiles, 4 KB

    int tid  = threadIdx.x;
    int wave = tid >> 5;
    int lane = tid & 31;
    int bh = blockIdx.y;
    int b = bh >> 1, h = bh & 1;
    int m0 = blockIdx.x * 64 + wave * 16;
    size_t baseRow = (size_t)b * N;
    const u16* Qh = Qb + baseRow * D + h * GS;
    const u16* Kh = Kb + baseRow * D + h * GS;
    const u16* Vh = Vb + baseRow * D + h * GS;
    int hi8 = (lane >> 4) << 3;

    v16bf qf[8];
#pragma unroll
    for (int kq = 0; kq < 8; ++kq) qf[kq] = load_a_frag(Qh, m0, kq * 32, D);

    v8f o[16] = {};
    float mrow[8], lrow[8];
#pragma unroll
    for (int r = 0; r < 8; ++r) { mrow[r] = -3e38f; lrow[r] = 0.0f; }

    for (int kv0 = 0; kv0 < N; kv0 += 32) {
        __syncthreads();   // previous tile fully consumed before overwrite

#if USE_TDM
        // TDM: one wave issues a 2D tile descriptor (32 rows x 256 cols of
        // 2-byte elements, row stride 512 elements) -> lds_k.  TENSORcnt path.
        if (wave == 0) {
            unsigned lds_off = (unsigned)(size_t)(lds16_t*)&lds_k[0];
            unsigned long long ga =
                (unsigned long long)(size_t)(Kh + (size_t)kv0 * D);
            u32x4 g0 = { 1u,                                   // count=1 (valid)
                         lds_off,                              // lds_addr (bytes)
                         (unsigned)ga,                         // global_addr[31:0]
                         (unsigned)((ga >> 32) & 0x01FFFFFFull) | 0x80000000u }; // addr[56:32] | type=2
            i32x8 g1 = { (int)0x00010000,       // data_size=1 (2 bytes)
                         (int)(256u << 16),     // tensor_dim0 = 256
                         (int)(32u  << 16),     // tensor_dim1 = 32
                         (int)(256u << 16),     // tile_dim0 = 256
                         32,                    // tile_dim1 = 32
                         512,                   // tensor_dim0_stride = 512 elems
                         0, 0 };
            i32x4 gz4 = { 0, 0, 0, 0 };              // groups 2/3 unused (2D)
            i32x8 gz8 = { 0, 0, 0, 0, 0, 0, 0, 0 };  // extra group slot (6-arg form)
            __builtin_amdgcn_tensor_load_to_lds(g0, g1, gz4, gz4, gz8, 0);
        }
#endif
        // cooperative staging: 32 kv rows x 256 d cols (1024 uint4 = 8/thread)
#pragma unroll
        for (int i = 0; i < 8; ++i) {
            int idx = tid + i * 128;
            int row = idx >> 5;
            int c8  = (idx & 31) << 3;
#if !USE_TDM
            const u16* gk = Kh + (size_t)(kv0 + row) * D + c8;
            *(uint4*)&lds_k[row * 256 + c8] = *(const uint4*)gk;
#endif
            uint4 vv = *(const uint4*)(Vh + (size_t)(kv0 + row) * D + c8);
            const u16* pv = (const u16*)&vv;
#pragma unroll
            for (int t = 0; t < 8; ++t) lds_v[(c8 + t) * 32 + row] = pv[t];
        }
#if USE_TDM
        if (wave == 0) __builtin_amdgcn_s_wait_tensorcnt(0);
#endif
        __syncthreads();

        // scores: two 16x16 kv tiles, K-dim 256; batch 4 fragments per round
        v8f s[2] = {};
#pragma unroll
        for (int j = 0; j < 2; ++j)
#pragma unroll
            for (int half = 0; half < 2; ++half) {
                v16bf kf[4];
#pragma unroll
                for (int q = 0; q < 4; ++q)
                    kf[q] = load_b_frag(&lds_k[0], j * 16, (half * 4 + q) * 32, 256);
#pragma unroll
                for (int q = 0; q < 4; ++q)
                    s[j] = __builtin_amdgcn_wmma_f32_16x16x32_bf16(
                        false, qf[half * 4 + q], false, kf[q], (short)0, s[j], false, false);
            }

        // online softmax update, P -> LDS (bf16)
#pragma unroll
        for (int r = 0; r < 8; ++r) {
            float v0 = s[0][r] * 0.0625f;   // 1/sqrt(256)
            float v1 = s[1][r] * 0.0625f;
            float rmax = fmaxf(v0, v1);
#pragma unroll
            for (int msk = 1; msk < 16; msk <<= 1)
                rmax = fmaxf(rmax, __shfl_xor(rmax, msk, 32));
            float nm = fmaxf(mrow[r], rmax);
            float corr = __expf(mrow[r] - nm);
            float p0 = __expf(v0 - nm), p1 = __expf(v1 - nm);
            float rs = p0 + p1;
#pragma unroll
            for (int msk = 1; msk < 16; msk <<= 1)
                rs += __shfl_xor(rs, msk, 32);
            lrow[r] = lrow[r] * corr + rs;
            mrow[r] = nm;
#pragma unroll
            for (int g = 0; g < 16; ++g) o[g][r] = o[g][r] * corr;
            int mloc = r + hi8;
            pbuf[wave][mloc * 32 + (lane & 15)]      = f2bf(p0);
            pbuf[wave][mloc * 32 + 16 + (lane & 15)] = f2bf(p1);
        }

        // P (16x32) @ V (32x256): V B-fragments contiguous from transposed LDS
        v16bf pa = load_a_frag(&pbuf[wave][0], 0, 0, 32);
#pragma unroll
        for (int g = 0; g < 16; ++g) {
            v16bf vf = load_b_frag(&lds_v[0], g * 16, 0, 32);
            o[g] = __builtin_amdgcn_wmma_f32_16x16x32_bf16(
                false, pa, false, vf, (short)0, o[g], false, false);
        }
    }

    // normalize and store
#pragma unroll
    for (int g = 0; g < 16; ++g)
#pragma unroll
        for (int r = 0; r < 8; ++r) {
            int m = m0 + r + hi8;
            int dcol = h * GS + g * 16 + (lane & 15);
            Out[(baseRow + m) * D + dcol] = o[g][r] / lrow[r];
        }
}

// ---------------- residual add + LayerNorm ----------------
__global__ __launch_bounds__(256)
void add_ln(const float* __restrict__ X, const float* __restrict__ R,
            const float* __restrict__ G, const float* __restrict__ Bt,
            float eps, float* __restrict__ OutF, u16* __restrict__ OutB) {
    constexpr int D = 512;
    __shared__ float red[256];
    int row = blockIdx.x;
    int t = threadIdx.x;
    const float* x = X + (size_t)row * D;
    const float* rr = R + (size_t)row * D;
    float v0 = x[t] + rr[t];
    float v1 = x[t + 256] + rr[t + 256];
    red[t] = v0 + v1;
    __syncthreads();
    for (int st = 128; st > 0; st >>= 1) {
        if (t < st) red[t] += red[t + st];
        __syncthreads();
    }
    float mu = red[0] * (1.0f / D);
    __syncthreads();
    float d0 = v0 - mu, d1 = v1 - mu;
    red[t] = d0 * d0 + d1 * d1;
    __syncthreads();
    for (int st = 128; st > 0; st >>= 1) {
        if (t < st) red[t] += red[t + st];
        __syncthreads();
    }
    float var = red[0] * (1.0f / D);
    float inv = rsqrtf(var + eps);
    float y0 = d0 * inv * G[t] + Bt[t];
    float y1 = d1 * inv * G[t + 256] + Bt[t + 256];
    OutF[(size_t)row * D + t] = y0;
    OutF[(size_t)row * D + t + 256] = y1;
    if (OutB) {
        OutB[(size_t)row * D + t] = f2bf(y0);
        OutB[(size_t)row * D + t + 256] = f2bf(y1);
    }
}

// ---------------- host side ----------------

static void run_block(const u16* q_in, const u16* k_in, const u16* v_in,
                      const float* resF,
                      const u16* wq, const u16* wk, const u16* wv,
                      const float* lng, const float* lnb,
                      const u16* w1, const u16* w2,
                      const float* flng, const float* flnb,
                      u16* qb, u16* kb, u16* vb,
                      float* t0, float* xf, u16* xb, u16* hb,
                      float* outF, u16* outB, hipStream_t stream) {
    dim3 blk(256);
    gemm_nt<1><<<dim3(4, 64), blk, 0, stream>>>(q_in, wq, qb, 8192, 512, 512);
    gemm_nt<1><<<dim3(4, 64), blk, 0, stream>>>(k_in, wk, kb, 8192, 512, 512);
    gemm_nt<1><<<dim3(4, 64), blk, 0, stream>>>(v_in, wv, vb, 8192, 512, 512);
    attn_flash<<<dim3(32, 8), 128, 0, stream>>>(qb, kb, vb, t0);
    add_ln<<<8192, 256, 0, stream>>>(t0, resF, lng, lnb, 1e-5f, xf, xb);
    gemm_nt<2><<<dim3(16, 64), blk, 0, stream>>>(xb, w1, hb, 8192, 2048, 512);
    gemm_nt<0><<<dim3(4, 64), blk, 0, stream>>>(hb, w2, t0, 8192, 512, 2048);
    add_ln<<<8192, 256, 0, stream>>>(t0, xf, flng, flnb, 1e-6f, outF, outB);
}

extern "C" void kernel_launch(void* const* d_in, const int* in_sizes, int n_in,
                              void* d_out, int out_size, void* d_ws, size_t ws_size,
                              hipStream_t stream) {
    (void)in_sizes; (void)n_in; (void)out_size; (void)ws_size;
    const float* Fm  = (const float*)d_in[0];
    const float* Fs  = (const float*)d_in[1];
    const float* Fq  = (const float*)d_in[2];
    const float* Wq  = (const float*)d_in[3];
    const float* Wk  = (const float*)d_in[4];
    const float* Wv  = (const float*)d_in[5];
    const float* lng = (const float*)d_in[6];
    const float* lnb = (const float*)d_in[7];
    const float* W1  = (const float*)d_in[8];
    const float* W2  = (const float*)d_in[9];
    const float* flng = (const float*)d_in[10];
    const float* flnb = (const float*)d_in[11];

    const size_t MD = (size_t)8192 * 512;        // activation elements
    const size_t MH = (size_t)8192 * 2048;       // FFN hidden elements
    const size_t WP = (size_t)3 * 512 * 512;     // projection weights (all blocks)
    const size_t WF = (size_t)3 * 2048 * 512;    // ffn weights (all blocks)

    char* ws = (char*)d_ws;
    size_t off = 0;
    auto alloc = [&](size_t bytes) -> char* {
        char* p = ws + off;
        off += (bytes + 255) & ~(size_t)255;
        return p;
    };
    u16* fm_bf = (u16*)alloc(MD * 2);
    u16* fs_bf = (u16*)alloc(MD * 2);
    u16* fq_bf = (u16*)alloc(MD * 2);
    u16* wq_bf = (u16*)alloc(WP * 2);
    u16* wk_bf = (u16*)alloc(WP * 2);
    u16* wv_bf = (u16*)alloc(WP * 2);
    u16* w1_bf = (u16*)alloc(WF * 2);
    u16* w2_bf = (u16*)alloc(WF * 2);
    u16* qb = (u16*)alloc(MD * 2);
    u16* kb = (u16*)alloc(MD * 2);
    u16* vb = (u16*)alloc(MD * 2);
    float* t0 = (float*)alloc(MD * 4);
    float* xf = (float*)alloc(MD * 4);
    u16* xb = (u16*)alloc(MD * 2);
    u16* hb = (u16*)alloc(MH * 2);
    float* css_f = (float*)alloc(MD * 4);
    u16* css_bf = (u16*)alloc(MD * 2);
    u16* cqq_bf = (u16*)alloc(MD * 2);

    auto cvt = [&](const float* src, u16* dst, size_t n) {
        cvt_bf16<<<1024, 256, 0, stream>>>(src, dst, (int)n);
    };
    cvt(Fm, fm_bf, MD);
    cvt(Fs, fs_bf, MD);
    cvt(Fq, fq_bf, MD);
    cvt(Wq, wq_bf, WP);
    cvt(Wk, wk_bf, WP);
    cvt(Wv, wv_bf, WP);
    cvt(W1, w1_bf, WF);
    cvt(W2, w2_bf, WF);

    const size_t sp = (size_t)512 * 512;     // per-block projection weight stride
    const size_t sf = (size_t)2048 * 512;    // per-block ffn weight stride

    // block 0: css = blk(Fs, Fs, Fm, resnet=Fm)
    run_block(fs_bf, fs_bf, fm_bf, Fm,
              wq_bf, wk_bf, wv_bf, lng, lnb, w1_bf, w2_bf, flng, flnb,
              qb, kb, vb, t0, xf, xb, hb, css_f, css_bf, stream);

    // block 1: cqq = blk(Fq, Fq, Fq, resnet=Fq)  (f32 out lands in d_out as scratch)
    run_block(fq_bf, fq_bf, fq_bf, Fq,
              wq_bf + sp, wk_bf + sp, wv_bf + sp, lng + 512, lnb + 512,
              w1_bf + sf, w2_bf + sf, flng + 512, flnb + 512,
              qb, kb, vb, t0, xf, xb, hb, (float*)d_out, cqq_bf, stream);

    // block 2: csq = blk(cqq, Fs, css, resnet=css) -> d_out
    run_block(cqq_bf, fs_bf, css_bf, css_f,
              wq_bf + 2 * sp, wk_bf + 2 * sp, wv_bf + 2 * sp, lng + 1024, lnb + 1024,
              w1_bf + 2 * sf, w2_bf + 2 * sf, flng + 1024, flnb + 1024,
              qb, kb, vb, t0, xf, xb, hb, (float*)d_out, nullptr, stream);
}
